// GAT_19971597926651
// MI455X (gfx1250) — matrix operand order repaired and checked by reading, not compiled, not run on hardware
//
#include <hip/hip_runtime.h>
#include <hip/hip_bf16.h>

// ---------- types ----------
typedef __attribute__((ext_vector_type(16))) __bf16    v16bf;
typedef __attribute__((ext_vector_type(8)))  float     v8f;
typedef __attribute__((ext_vector_type(4)))  unsigned  u32x4;   // POD vector: addrspace-safe

// global-address-space views (force global_load/global_store)
typedef __attribute__((address_space(1))) const u32x4  g_cu32x4;
typedef __attribute__((address_space(1))) float        g_float;

__device__ __forceinline__ g_cu32x4* gp4(const void* p) {
    return (g_cu32x4*)(unsigned long long)p;
}
__device__ __forceinline__ g_float* gpf(float* p) {
    return (g_float*)(unsigned long long)p;
}

#define F_DIM 256
#define HEADS 8
#define HDIM  32

// ---------- helpers ----------
__device__ __forceinline__ unsigned short f2bf(float f) {
    unsigned u = __builtin_bit_cast(unsigned, f);
    unsigned r = u + 0x7fffu + ((u >> 16) & 1u);   // round-to-nearest-even
    return (unsigned short)(r >> 16);
}
__device__ __forceinline__ unsigned fkey(float f) {   // order-preserving float->uint
    unsigned u = __builtin_bit_cast(unsigned, f);
    return (u & 0x80000000u) ? ~u : (u | 0x80000000u);
}
__device__ __forceinline__ float funkey(unsigned k) {
    unsigned u = (k & 0x80000000u) ? (k & 0x7fffffffu) : ~k;
    return __builtin_bit_cast(float, u);
}
__device__ __forceinline__ v16bf mkfrag(u32x4 a, u32x4 b) {
    union { u32x4 u[2]; v16bf v; } t;
    t.u[0] = a; t.u[1] = b;
    return t.v;
}
__device__ __forceinline__ v8f wmma_bf16(v16bf a, v16bf b, v8f c) {
    return __builtin_amdgcn_wmma_f32_16x16x32_bf16(false, a, false, b, (short)0, c, false, false);
}

// ---------- pack kernels ----------
__global__ void gat_f2bf_kernel(const float* __restrict__ in, unsigned short* __restrict__ out, int n) {
    int t = blockIdx.x * blockDim.x + threadIdx.x;
    if (t < n) out[t] = f2bf(in[t]);
}

// Pack weight W[K][NC] (row-major f32) into WMMA B-fragment order:
// out[((ct*KS + ks)*32 + lane)*8 + v] = {lo: bf16(W[k0][n]), hi: bf16(W[k0+1][n])}
// n = ct*16 + lane%16, k0 = ks*32 + (v/4)*16 + (lane/16)*8 + 2*(v%4)   (ISA 16-bit B 32x16 layout)
__global__ void gat_packB_kernel(const float* __restrict__ W, unsigned* __restrict__ out, int K, int NC) {
    int t = blockIdx.x * blockDim.x + threadIdx.x;
    int total = (K * NC) >> 1;
    if (t >= total) return;
    int v    = t & 7;
    int lane = (t >> 3) & 31;
    int rest = t >> 8;
    int KS   = K >> 5;
    int ks   = rest % KS;
    int ct   = rest / KS;
    int n  = ct * 16 + (lane & 15);
    int k0 = ks * 32 + (v >> 2) * 16 + (lane >> 4) * 8 + 2 * (v & 3);
    unsigned lo = f2bf(W[(size_t)k0 * NC + n]);
    unsigned hi = f2bf(W[(size_t)(k0 + 1) * NC + n]);
    out[t] = lo | (hi << 16);
}

// ---------- 4-chain K=256 GEMM step with register-cached A + pipelined B ----------
// 4 independent WMMA chains cover the bf16 WMMA->WMMA hazard window (1+4 co-exec)
// and software-pipelined B loads keep LOADcnt from draining before each WMMA.
__device__ __forceinline__ void gemm4_regA(const u32x4 a0[8], const u32x4 a1[8],
                                           const unsigned* __restrict__ pw, int ct, int lane,
                                           v8f acc[4]) {
    u32x4 cur[8], nxt[8];
#pragma unroll
    for (int c = 0; c < 4; ++c) {
        g_cu32x4* bp = gp4(pw + ((size_t)(((ct + c) * 8 + 0) * 32)) * 8) + lane * 2;
        cur[2 * c] = bp[0]; cur[2 * c + 1] = bp[1];
    }
#pragma unroll
    for (int ks = 0; ks < 8; ++ks) {
        if (ks < 7) {
#pragma unroll
            for (int c = 0; c < 4; ++c) {
                g_cu32x4* bp = gp4(pw + ((size_t)(((ct + c) * 8 + ks + 1) * 32)) * 8) + lane * 2;
                nxt[2 * c] = bp[0]; nxt[2 * c + 1] = bp[1];
            }
        }
        v16bf av = mkfrag(a0[ks], a1[ks]);
#pragma unroll
        for (int c = 0; c < 4; ++c)
            acc[c] = wmma_bf16(av, mkfrag(cur[2 * c], cur[2 * c + 1]), acc[c]);
#pragma unroll
        for (int i = 0; i < 8; ++i) cur[i] = nxt[i];
    }
}

// ---------- one 16-row x 256-col GEMM against a packed weight ----------
__device__ __forceinline__ void gemm_tile_256(const u32x4 a0[8], const u32x4 a1[8],
                                              const unsigned* __restrict__ pw,
                                              float* __restrict__ ow,
                                              int m0, int lm, int lh) {
    int lane = threadIdx.x;
    for (int ct = 0; ct < 16; ct += 4) {
        v8f acc[4] = {};
        gemm4_regA(a0, a1, pw, ct, lane, acc);
#pragma unroll
        for (int r = 0; r < 8; ++r) {
            size_t row = (size_t)(m0 + r + lh * 8) * F_DIM;
#pragma unroll
            for (int c = 0; c < 4; ++c)
                *gpf(ow + row + (ct + c) * 16 + lm) = acc[c][r];
        }
    }
}

// ---------- QKV projection: one wave per 16-row tile ----------
__global__ void gat_qkv_kernel(const unsigned short* __restrict__ featb,
                               const unsigned* __restrict__ pq,
                               const unsigned* __restrict__ pk,
                               const unsigned* __restrict__ pv,
                               float* __restrict__ q, float* __restrict__ k, float* __restrict__ v,
                               int ntiles) {
    int tile = blockIdx.x * blockDim.y + threadIdx.y;
    if (tile >= ntiles) return;
    int lane = threadIdx.x, lm = lane & 15, lh = lane >> 4;
    int m0 = tile * 16;

    // Cache 8 A fragments (K=256): two global b128 loads each (row-major bf16)
    u32x4 a0[8], a1[8];
    const unsigned short* arow = featb + (size_t)(m0 + lm) * F_DIM + lh * 8;
#pragma unroll
    for (int ks = 0; ks < 8; ++ks) {
        g_cu32x4* p = gp4(arow + ks * 32);
        a0[ks] = p[0];
        a1[ks] = p[2];   // +16 elements
    }
    gemm_tile_256(a0, a1, pq, q, m0, lm, lh);
    gemm_tile_256(a0, a1, pk, k, m0, lm, lh);
    gemm_tile_256(a0, a1, pv, v, m0, lm, lh);
}

// ---------- edge scores + segment max ----------
__global__ void gat_scores_kernel(const float* __restrict__ q, const float* __restrict__ k,
                                  const int* __restrict__ esrc, const int* __restrict__ edst,
                                  float* __restrict__ e, unsigned* __restrict__ mkey, long long EH) {
    long long t = blockIdx.x * (long long)blockDim.x + threadIdx.x;
    if (t >= EH) return;
    int h = (int)(t & 7);
    long long ei = t >> 3;
    int s = esrc[ei], d = edst[ei];
    const float4* kp = (const float4*)(k + (size_t)s * F_DIM + h * HDIM);
    const float4* qp = (const float4*)(q + (size_t)d * F_DIM + h * HDIM);
    float acc = 0.f;
#pragma unroll
    for (int i = 0; i < 8; ++i) {
        float4 a = kp[i], b = qp[i];
        acc += a.x * b.x + a.y * b.y + a.z * b.z + a.w * b.w;
    }
    acc *= 0.0625f;  // / sqrt(256)
    e[t] = acc;
    atomicMax(&mkey[(size_t)d * HEADS + h], fkey(acc));
}

// ---------- p = exp(e - m), z = segment_sum(p) ----------
__global__ void gat_expsum_kernel(float* __restrict__ e, const unsigned* __restrict__ mkey,
                                  float* __restrict__ z, const int* __restrict__ edst, long long EH) {
    long long t = blockIdx.x * (long long)blockDim.x + threadIdx.x;
    if (t >= EH) return;
    int h = (int)(t & 7);
    long long ei = t >> 3;
    int d = edst[ei];
    float m = funkey(mkey[(size_t)d * HEADS + h]);
    float p = __expf(e[t] - m);
    e[t] = p;
    atomicAdd(&z[(size_t)d * HEADS + h], p);
}

// ---------- weighted scatter-sum of messages ----------
__global__ void gat_agg_kernel(const float* __restrict__ v, const float* __restrict__ p,
                               const float* __restrict__ z, const int* __restrict__ esrc,
                               const int* __restrict__ edst, float* __restrict__ ft2, long long EC) {
    long long t = blockIdx.x * (long long)blockDim.x + threadIdx.x;
    if (t >= EC) return;
    int c = (int)(t & 255);
    long long ei = t >> 8;
    int h = c >> 5;
    int s = esrc[ei], d = edst[ei];
    float a = p[ei * HEADS + h] / z[(size_t)d * HEADS + h];
    atomicAdd(&ft2[(size_t)d * F_DIM + c], v[(size_t)s * F_DIM + c] * a);
}

// ---------- LN1: rst = LN(ft2 + feat); also emit bf16 copy ----------
__global__ void gat_ln1_kernel(const float* __restrict__ ft2, const float* __restrict__ feat,
                               const float* __restrict__ g, const float* __restrict__ b,
                               float* __restrict__ rst, unsigned short* __restrict__ rstb, int N) {
    int row = blockIdx.x * blockDim.y + threadIdx.y;
    if (row >= N) return;
    int lane = threadIdx.x;
    const float* xa = ft2  + (size_t)row * F_DIM;
    const float* xb = feat + (size_t)row * F_DIM;
    float xv[8];
    float s = 0.f, s2 = 0.f;
#pragma unroll
    for (int j = 0; j < 8; ++j) {
        int c = lane * 8 + j;
        float x = xa[c] + xb[c];
        xv[j] = x;
        s += x;
        s2 += x * x;
    }
#pragma unroll
    for (int off = 16; off; off >>= 1) {
        s  += __shfl_xor(s, off);
        s2 += __shfl_xor(s2, off);
    }
    float mu = s * (1.0f / F_DIM);
    float var = s2 * (1.0f / F_DIM) - mu * mu;
    float rinv = rsqrtf(var + 1e-5f);
#pragma unroll
    for (int j = 0; j < 8; ++j) {
        int c = lane * 8 + j;
        float y = (xv[j] - mu) * rinv * g[c] + b[c];
        rst[(size_t)row * F_DIM + c]  = y;
        rstb[(size_t)row * F_DIM + c] = f2bf(y);
    }
}

// ---------- fused FFN + LN2: one wave per 16-row tile ----------
__global__ void gat_ffn_kernel(const unsigned short* __restrict__ rstb,
                               const float* __restrict__ rst,
                               const unsigned* __restrict__ pw1, const float* __restrict__ b1,
                               const float* __restrict__ alpha,
                               const unsigned* __restrict__ pw2, const float* __restrict__ b2,
                               const float* __restrict__ g, const float* __restrict__ bb,
                               float* __restrict__ out, int ntiles) {
    __shared__ unsigned short hbuf[16 * 1024]; // 32 KB: 16x1024 bf16 hidden tile (row-major)
    __shared__ float          obuf[16 * 256];  // 16 KB: rst + ffn tile

    int tile = blockIdx.x;
    if (tile >= ntiles) return;
    int lane = threadIdx.x, lm = lane & 15, lh = lane >> 4;
    int m0 = tile * 16;

    // Cache A fragments of rst tile (K=256 -> 8 steps)
    u32x4 a0[8], a1[8];
    const unsigned short* arow = rstb + (size_t)(m0 + lm) * F_DIM + lh * 8;
#pragma unroll
    for (int ks = 0; ks < 8; ++ks) {
        g_cu32x4* p = gp4(arow + ks * 32);
        a0[ks] = p[0];
        a1[ks] = p[2];
    }

    // Phase 1: h = PReLU(rst @ W1 + b1)  (16 x 1024) -> LDS as bf16
    for (int ct = 0; ct < 64; ct += 4) {
        v8f acc[4] = {};
        gemm4_regA(a0, a1, pw1, ct, lane, acc);
#pragma unroll
        for (int r = 0; r < 8; ++r) {
            int row = r + lh * 8;
#pragma unroll
            for (int c = 0; c < 4; ++c) {
                int col = (ct + c) * 16 + lm;
                float hv = acc[c][r] + b1[col];
                hv = hv > 0.f ? hv : alpha[col] * hv;
                hbuf[row * 1024 + col] = f2bf(hv);
            }
        }
    }
    __syncthreads();

    // Phase 2: ffn = h @ W2 + b2; obuf = rst + ffn  (16 x 256)
    // 4 chains: one LDS A fragment feeds 4 WMMAs; B pipelined from global.
    for (int ct = 0; ct < 16; ct += 4) {
        v8f acc[4] = {};
        u32x4 cur[8], nxt[8];
#pragma unroll
        for (int c = 0; c < 4; ++c) {
            g_cu32x4* bp = gp4(pw2 + ((size_t)(((ct + c) * 32 + 0) * 32)) * 8) + lane * 2;
            cur[2 * c] = bp[0]; cur[2 * c + 1] = bp[1];
        }
        for (int ks = 0; ks < 32; ++ks) {
            if (ks < 31) {
#pragma unroll
                for (int c = 0; c < 4; ++c) {
                    g_cu32x4* bp = gp4(pw2 + ((size_t)(((ct + c) * 32 + ks + 1) * 32)) * 8) + lane * 2;
                    nxt[2 * c] = bp[0]; nxt[2 * c + 1] = bp[1];
                }
            }
            const u32x4* ap = (const u32x4*)(&hbuf[lm * 1024 + ks * 32 + lh * 8]);
            v16bf av = mkfrag(ap[0], ap[2]);
#pragma unroll
            for (int c = 0; c < 4; ++c)
                acc[c] = wmma_bf16(av, mkfrag(cur[2 * c], cur[2 * c + 1]), acc[c]);
#pragma unroll
            for (int i = 0; i < 8; ++i) cur[i] = nxt[i];
        }
#pragma unroll
        for (int r = 0; r < 8; ++r) {
            int row = r + lh * 8;
#pragma unroll
            for (int c = 0; c < 4; ++c) {
                int col = (ct + c) * 16 + lm;
                obuf[row * 256 + col] = acc[c][r] + b2[col] + rst[(size_t)(m0 + row) * F_DIM + col];
            }
        }
    }
    __syncthreads();

    // LN2: 2 lanes per row, 128 cols each
    {
        int row = lane >> 1;
        int half = lane & 1;
        const float* xr = &obuf[row * 256 + half * 128];
        float s = 0.f, s2 = 0.f;
        for (int j = 0; j < 128; ++j) {
            float x = xr[j];
            s += x;
            s2 += x * x;
        }
        s  += __shfl_xor(s, 1);
        s2 += __shfl_xor(s2, 1);
        float mu = s * (1.0f / F_DIM);
        float var = s2 * (1.0f / F_DIM) - mu * mu;
        float rinv = rsqrtf(var + 1e-5f);
        for (int j = 0; j < 128; ++j) {
            int c = half * 128 + j;
            *gpf(out + (size_t)(m0 + row) * F_DIM + c) = (xr[j] - mu) * rinv * g[c] + bb[c];
        }
    }
}

// ---------- host ----------
extern "C" void kernel_launch(void* const* d_in, const int* in_sizes, int n_in,
                              void* d_out, int out_size, void* d_ws, size_t ws_size,
                              hipStream_t stream) {
    const float* feat   = (const float*)d_in[0];
    const int*   esrc   = (const int*)d_in[1];
    const int*   edst   = (const int*)d_in[2];
    const float* Wq     = (const float*)d_in[3];
    const float* Wk     = (const float*)d_in[4];
    const float* Wv     = (const float*)d_in[5];
    const float* ln_g   = (const float*)d_in[6];
    const float* ln_b   = (const float*)d_in[7];
    const float* W1     = (const float*)d_in[8];
    const float* b1     = (const float*)d_in[9];
    const float* alpha  = (const float*)d_in[10];
    const float* W2     = (const float*)d_in[11];
    const float* b2     = (const float*)d_in[12];

    const int N = in_sizes[0] / F_DIM;
    const long long E = in_sizes[1];

    // carve workspace (256B aligned)
    char* w = (char*)d_ws;
    size_t off = 0;
    auto alloc = [&](size_t bytes) -> void* {
        off = (off + 255) & ~(size_t)255;
        void* p = w + off;
        off += bytes;
        return p;
    };
    unsigned short* featb = (unsigned short*)alloc((size_t)N * F_DIM * 2);
    unsigned*       pWq   = (unsigned*)alloc((size_t)F_DIM * F_DIM * 2);
    unsigned*       pWk   = (unsigned*)alloc((size_t)F_DIM * F_DIM * 2);
    unsigned*       pWv   = (unsigned*)alloc((size_t)F_DIM * F_DIM * 2);
    unsigned*       pW1   = (unsigned*)alloc((size_t)F_DIM * 1024 * 2);
    unsigned*       pW2   = (unsigned*)alloc((size_t)1024 * F_DIM * 2);
    float*          q     = (float*)alloc((size_t)N * F_DIM * 4);
    float*          k     = (float*)alloc((size_t)N * F_DIM * 4);
    float*          v     = (float*)alloc((size_t)N * F_DIM * 4);
    float*          e     = (float*)alloc((size_t)E * HEADS * 4);
    unsigned*       mkey  = (unsigned*)alloc((size_t)N * HEADS * 4);
    float*          z     = (float*)alloc((size_t)N * HEADS * 4);
    float*          ft2   = (float*)alloc((size_t)N * F_DIM * 4);
    float*          rst   = (float*)alloc((size_t)N * F_DIM * 4);
    unsigned short* rstb  = (unsigned short*)alloc((size_t)N * F_DIM * 2);
    (void)ws_size; (void)n_in; (void)out_size;

    (void)hipMemsetAsync(mkey, 0, (size_t)N * HEADS * 4, stream);
    (void)hipMemsetAsync(z,    0, (size_t)N * HEADS * 4, stream);
    (void)hipMemsetAsync(ft2,  0, (size_t)N * F_DIM * 4, stream);

    // pack feat + weights into bf16 / fragment order
    {
        int n = N * F_DIM;
        gat_f2bf_kernel<<<(n + 255) / 256, 256, 0, stream>>>(feat, featb, n);
        int tqkv = (F_DIM * F_DIM) / 2;
        gat_packB_kernel<<<(tqkv + 255) / 256, 256, 0, stream>>>(Wq, pWq, F_DIM, F_DIM);
        gat_packB_kernel<<<(tqkv + 255) / 256, 256, 0, stream>>>(Wk, pWk, F_DIM, F_DIM);
        gat_packB_kernel<<<(tqkv + 255) / 256, 256, 0, stream>>>(Wv, pWv, F_DIM, F_DIM);
        int t1 = (F_DIM * 1024) / 2;
        gat_packB_kernel<<<(t1 + 255) / 256, 256, 0, stream>>>(W1, pW1, F_DIM, 1024);
        int t2 = (1024 * F_DIM) / 2;
        gat_packB_kernel<<<(t2 + 255) / 256, 256, 0, stream>>>(W2, pW2, 1024, F_DIM);
    }

    const int ntiles = N / 16;  // N = 20000 divisible by 16
    {
        dim3 blk(32, 8);
        dim3 grd((ntiles + 7) / 8);
        gat_qkv_kernel<<<grd, blk, 0, stream>>>(featb, pWq, pWk, pWv, q, k, v, ntiles);
    }
    {
        long long EH = E * HEADS;
        gat_scores_kernel<<<(unsigned)((EH + 255) / 256), 256, 0, stream>>>(q, k, esrc, edst, e, mkey, EH);
        gat_expsum_kernel<<<(unsigned)((EH + 255) / 256), 256, 0, stream>>>(e, mkey, z, edst, EH);
        long long EC = E * F_DIM;
        gat_agg_kernel<<<(unsigned)((EC + 255) / 256), 256, 0, stream>>>(v, e, z, esrc, edst, ft2, EC);
    }
    {
        dim3 blk(32, 8);
        dim3 grd((N + 7) / 8);
        gat_ln1_kernel<<<grd, blk, 0, stream>>>(ft2, feat, ln_g, ln_b, rst, rstb, N);
    }
    {
        gat_ffn_kernel<<<ntiles, 32, 0, stream>>>(rstb, rst, pW1, b1, alpha, pW2, b2,
                                                  ln_g, ln_b, (float*)d_out, ntiles);
    }
}
